// GENNet_28836410425878
// MI455X (gfx1250) — matrix operand order repaired and checked by reading, not compiled, not run on hardware
//
#include <hip/hip_runtime.h>

// ---------------------------------------------------------------------------
// GENNet (2x GENConv softmax-agg + MLP/BN, mean-pool, FC) for gfx1250.
//   - edge softmax aggregation: int-bits atomic max (all msgs > 0) + fp32
//     atomic adds; node features (25.6 MB) are L2-resident on MI455X (192 MB).
//   - GEMMs via v_wmma_f32_16x16x32_f16, one 16x16 f32 tile per wave32.
//     Weights staged transposed in LDS as f16 (16 KB/block) so B fragments
//     are contiguous ds_load_b128s; bias folded into accumulator init,
//     BN+ReLU fused into A-fragment loads.
// ---------------------------------------------------------------------------

typedef __attribute__((ext_vector_type(16))) _Float16 v16h;
typedef __attribute__((ext_vector_type(8)))  float    v8f;

#define MSG_EPS 1e-7f
#define THREADS 256

// ---------------------------------- utility --------------------------------

__global__ void k_zero(float* __restrict__ p, long long n) {
  long long i = (long long)blockIdx.x * blockDim.x + threadIdx.x;
  long long stride = (long long)gridDim.x * blockDim.x;
  for (; i < n; i += stride) p[i] = 0.0f;
}

// ------------------------------- edge kernels ------------------------------
// msg = relu(x[src]) + eps ; segment max over dst via atomic int-max
__global__ void k_edge_max(const float* __restrict__ X,
                           const long long* __restrict__ ei,
                           float* __restrict__ m, int E) {
  int idx = blockIdx.x * blockDim.x + threadIdx.x;
  if (idx >= E * 16) return;
  int e = idx >> 4, q = idx & 15;
  int src = (int)ei[e];
  int dst = (int)ei[(size_t)E + e];
  const float* xr = X + (size_t)src * 64 + q * 4;
  __builtin_prefetch(xr, 0, 0);                    // global_prefetch_b8
  float4 v = *(const float4*)xr;
  float a0 = fmaxf(v.x, 0.f) + MSG_EPS;
  float a1 = fmaxf(v.y, 0.f) + MSG_EPS;
  float a2 = fmaxf(v.z, 0.f) + MSG_EPS;
  float a3 = fmaxf(v.w, 0.f) + MSG_EPS;
  int* mp = (int*)(m + (size_t)dst * 64 + q * 4);  // msgs > 0: int cmp == fp cmp
  atomicMax(mp + 0, __float_as_int(a0));
  atomicMax(mp + 1, __float_as_int(a1));
  atomicMax(mp + 2, __float_as_int(a2));
  atomicMax(mp + 3, __float_as_int(a3));
}

// ex = exp(msg - m[dst]); s1 += ex; s2 += ex*msg   (fp32 VMEM atomics)
__global__ void k_edge_sum(const float* __restrict__ X,
                           const long long* __restrict__ ei,
                           const float* __restrict__ m,
                           float* __restrict__ s1, float* __restrict__ s2,
                           int E) {
  int idx = blockIdx.x * blockDim.x + threadIdx.x;
  if (idx >= E * 16) return;
  int e = idx >> 4, q = idx & 15;
  int src = (int)ei[e];
  int dst = (int)ei[(size_t)E + e];
  const float* xr = X + (size_t)src * 64 + q * 4;
  __builtin_prefetch(xr, 0, 0);
  float4 v = *(const float4*)xr;
  size_t o = (size_t)dst * 64 + q * 4;
  float4 mv = *(const float4*)(m + o);
  float msg[4] = { fmaxf(v.x,0.f)+MSG_EPS, fmaxf(v.y,0.f)+MSG_EPS,
                   fmaxf(v.z,0.f)+MSG_EPS, fmaxf(v.w,0.f)+MSG_EPS };
  float mm[4] = { mv.x, mv.y, mv.z, mv.w };
#pragma unroll
  for (int j = 0; j < 4; ++j) {
    float ex = __expf(msg[j] - mm[j]);
    atomicAdd(s1 + o + j, ex);
    atomicAdd(s2 + o + j, ex * msg[j]);
  }
}

// xres = agg + x  (in place over s2)
__global__ void k_node(const float* __restrict__ X, const float* __restrict__ s1,
                       float* __restrict__ s2, long long n) {
  long long i = (long long)blockIdx.x * blockDim.x + threadIdx.x;
  if (i >= n) return;
  s2[i] = s2[i] / (s1[i] + 1e-16f) + X[i];
}

// ------------------------------- WMMA GEMMs --------------------------------
// H = xres[N,64] @ Wa[64,128] + ba.  Block = 8 waves = one 16-row stripe x
// all 8 column tiles; Wa staged transposed ([n][k], f16) in LDS.
__global__ void k_gemm_h(const float* __restrict__ A, const float* __restrict__ W,
                         const float* __restrict__ bias, float* __restrict__ Hout,
                         int Nrows) {
  __shared__ _Float16 WT[128 * 64];                 // WT[n*64 + k], 16 KB
  for (int t = threadIdx.x; t < 64 * 128; t += THREADS) {
    int k = t >> 7, n = t & 127;                    // W row-major [64,128]
    WT[n * 64 + k] = (_Float16)W[t];
  }
  __syncthreads();

  int wave = threadIdx.x >> 5, lane = threadIdx.x & 31;
  int gwave = blockIdx.x * 8 + wave;
  int rowTiles = (Nrows + 15) >> 4;
  if (gwave >= rowTiles * 8) return;
  int rt = gwave >> 3, ct = gwave & 7;
  if (rt * 16 + 16 > Nrows) return;
  int l15 = lane & 15;
  bool hi = lane >= 16;
  int row = rt * 16 + l15;
  int col = ct * 16 + l15;
  int lo = hi ? 16 : 0, hb = hi ? 24 : 8, kb = hi ? 16 : 0;

  v8f c;
  float bv = bias[col];
#pragma unroll
  for (int v = 0; v < 8; ++v) c[v] = bv;

#pragma unroll
  for (int kk = 0; kk < 64; kk += 32) {
    v16h a, b;
    const float* ar = A + (size_t)row * 64 + kk;
#pragma unroll
    for (int e = 0; e < 8; ++e) {
      a[e]     = (_Float16)ar[lo + e];
      a[8 + e] = (_Float16)ar[hb + e];
    }
    const _Float16* bp = WT + col * 64 + kk + kb;   // contiguous along K
#pragma unroll
    for (int e = 0; e < 16; ++e) b[e] = bp[e];      // -> ds_load_b128 x2
    c = __builtin_amdgcn_wmma_f32_16x16x32_f16(false, a, false, b,
                                               (short)0, c, false, false);
  }
#pragma unroll
  for (int v = 0; v < 8; ++v)
    Hout[(size_t)(rt * 16 + v + (hi ? 8 : 0)) * 128 + col] = c[v];
}

// Y = relu( relu(scale*H+shift)[N,128] @ Wb[128,64] + bb ).  Wb + BN vectors
// staged in LDS; block = 8 waves = two 16-row stripes x 4 column tiles.
__global__ void k_gemm_out(const float* __restrict__ Hm,
                           const float* __restrict__ scale,
                           const float* __restrict__ shift,
                           const float* __restrict__ W,
                           const float* __restrict__ bias,
                           float* __restrict__ Y, int Nrows) {
  __shared__ _Float16 WT[64 * 128];                 // WT[n*128 + k], 16 KB
  __shared__ float sc[128], sh[128];
  for (int t = threadIdx.x; t < 128 * 64; t += THREADS) {
    int k = t >> 6, n = t & 63;                     // W row-major [128,64]
    WT[n * 128 + k] = (_Float16)W[t];
  }
  if (threadIdx.x < 128) {
    sc[threadIdx.x] = scale[threadIdx.x];
    sh[threadIdx.x] = shift[threadIdx.x];
  }
  __syncthreads();

  int wave = threadIdx.x >> 5, lane = threadIdx.x & 31;
  int gwave = blockIdx.x * 8 + wave;
  int rowTiles = (Nrows + 15) >> 4;
  if (gwave >= rowTiles * 4) return;
  int rt = gwave >> 2, ct = gwave & 3;
  if (rt * 16 + 16 > Nrows) return;
  int l15 = lane & 15;
  bool hi = lane >= 16;
  int row = rt * 16 + l15;
  int col = ct * 16 + l15;
  int lo = hi ? 16 : 0, hb = hi ? 24 : 8, kb = hi ? 16 : 0;

  v8f c;
  float bv = bias[col];
#pragma unroll
  for (int v = 0; v < 8; ++v) c[v] = bv;

  const float* hr = Hm + (size_t)row * 128;
#pragma unroll
  for (int kk = 0; kk < 128; kk += 32) {
    v16h a, b;
#pragma unroll
    for (int e = 0; e < 8; ++e) {
      int k0 = kk + lo + e, k1 = kk + hb + e;
      a[e]     = (_Float16)fmaxf(fmaf(hr[k0], sc[k0], sh[k0]), 0.f);
      a[8 + e] = (_Float16)fmaxf(fmaf(hr[k1], sc[k1], sh[k1]), 0.f);
    }
    const _Float16* bp = WT + col * 128 + kk + kb;  // contiguous along K
#pragma unroll
    for (int e = 0; e < 16; ++e) b[e] = bp[e];
    c = __builtin_amdgcn_wmma_f32_16x16x32_f16(false, a, false, b,
                                               (short)0, c, false, false);
  }
#pragma unroll
  for (int v = 0; v < 8; ++v)
    Y[(size_t)(rt * 16 + v + (hi ? 8 : 0)) * 64 + col] = fmaxf(c[v], 0.f);
}

// ------------------------------ BatchNorm stats ----------------------------
__global__ void k_bnstat(const float* __restrict__ Hm, float* __restrict__ sum,
                         float* __restrict__ sumsq, int Nrows) {
  int c = threadIdx.x;  // 128
  float s = 0.f, sq = 0.f;
  for (int r = blockIdx.x; r < Nrows; r += gridDim.x) {
    float v = Hm[(size_t)r * 128 + c];
    s += v; sq += v * v;
  }
  atomicAdd(sum + c, s);
  atomicAdd(sumsq + c, sq);
}

__global__ void k_bnfinal(const float* __restrict__ sum,
                          const float* __restrict__ sumsq,
                          const float* __restrict__ gamma,
                          const float* __restrict__ beta,
                          float* __restrict__ scale, float* __restrict__ shift,
                          int Nrows) {
  int c = threadIdx.x;
  float invN = 1.0f / (float)Nrows;
  float mu  = sum[c] * invN;
  float var = sumsq[c] * invN - mu * mu;
  float s   = gamma[c] * rsqrtf(var + 1e-5f);
  scale[c] = s;
  shift[c] = beta[c] - mu * s;
}

// ------------------------------- pool + FC ---------------------------------
__global__ void k_pool(const float* __restrict__ Y, const long long* __restrict__ batch,
                       float* __restrict__ pooled, float* __restrict__ cnt,
                       long long n64) {
  long long idx = (long long)blockIdx.x * blockDim.x + threadIdx.x;
  if (idx >= n64) return;
  long long i = idx >> 6;
  int f = (int)(idx & 63);
  int g = (int)batch[i];
  atomicAdd(pooled + (size_t)g * 64 + f, Y[idx]);
  if (f == 0) atomicAdd(cnt + g, 1.0f);
}

__global__ void k_fc(const float* __restrict__ pooled, const float* __restrict__ cnt,
                     const float* __restrict__ Wfc, const float* __restrict__ bfc,
                     float* __restrict__ out, int Gn) {
  int g = threadIdx.x;
  if (g >= Gn) return;
  float inv = 1.0f / fmaxf(cnt[g], 1.0f);
#pragma unroll
  for (int c = 0; c < 10; ++c) {
    float acc = bfc[c];
    for (int k = 0; k < 64; ++k)
      acc = fmaf(pooled[(size_t)g * 64 + k] * inv, Wfc[k * 10 + c], acc);
    out[g * 10 + c] = acc;
  }
}

// --------------------------------- launcher --------------------------------
extern "C" void kernel_launch(void* const* d_in, const int* in_sizes, int n_in,
                              void* d_out, int out_size, void* d_ws, size_t ws_size,
                              hipStream_t stream) {
  const float*     x     = (const float*)d_in[0];
  const long long* ei    = (const long long*)d_in[1];
  const long long* batch = (const long long*)d_in[2];
  const float* Wa1 = (const float*)d_in[3];
  const float* ba1 = (const float*)d_in[4];
  const float* ga1 = (const float*)d_in[5];
  const float* be1 = (const float*)d_in[6];
  const float* Wb1 = (const float*)d_in[7];
  const float* bb1 = (const float*)d_in[8];
  const float* Wa2 = (const float*)d_in[9];
  const float* ba2 = (const float*)d_in[10];
  const float* ga2 = (const float*)d_in[11];
  const float* be2 = (const float*)d_in[12];
  const float* Wb2 = (const float*)d_in[13];
  const float* bb2 = (const float*)d_in[14];
  const float* Wfc = (const float*)d_in[15];
  const float* bfc = (const float*)d_in[16];
  float* out = (float*)d_out;

  const int N  = in_sizes[0] / 64;
  const int E  = in_sizes[1] / 2;
  const int Gn = out_size / 10;

  float* ws = (float*)d_ws;
  const size_t NF = (size_t)N * 64;
  float* m_buf = ws;            // [N,64] seg-max          (reused as act2)
  float* s1    = ws + 1 * NF;   // [N,64] softmax denom
  float* s2    = ws + 2 * NF;   // [N,64] weighted sum -> xres (in place)
  float* hbuf  = ws + 3 * NF;   // [N,128] hidden
  float* act1  = ws + 5 * NF;   // [N,64] layer-1 output
  float* S     = ws + 6 * NF;   // small region
  float* bnsum   = S;
  float* bnsumsq = S + 128;
  float* scale   = S + 256;
  float* shift   = S + 384;
  float* pooled  = S + 512;                 // [Gn,64]
  float* cnt     = pooled + (size_t)Gn * 64;
  float* act2 = m_buf;                      // m dead before gemm_out(layer2)

  const long long n64  = (long long)N * 64;
  const int gN64  = (int)((n64 + THREADS - 1) / THREADS);
  const int gN64x3 = (int)((3 * n64 + THREADS - 1) / THREADS);
  const int gE16  = (E * 16 + THREADS - 1) / THREADS;
  const int rowT  = (N + 15) / 16;
  const int gG1   = rowT;                       // 8 waves = 8 col-tiles/stripe
  const int gG2   = (rowT * 4 + 7) / 8;

  auto layer = [&](const float* Xin, const float* Wa, const float* ba,
                   const float* ga, const float* be, const float* Wb,
                   const float* bb, float* Yout) {
    k_zero<<<gN64x3, THREADS, 0, stream>>>(m_buf, 3 * n64);   // m,s1,s2 contig
    k_edge_max<<<gE16, THREADS, 0, stream>>>(Xin, ei, m_buf, E);
    k_edge_sum<<<gE16, THREADS, 0, stream>>>(Xin, ei, m_buf, s1, s2, E);
    k_node<<<gN64, THREADS, 0, stream>>>(Xin, s1, s2, n64);
    k_gemm_h<<<gG1, THREADS, 0, stream>>>(s2, Wa, ba, hbuf, N);
    k_zero<<<1, 512, 0, stream>>>(S, 512);
    k_bnstat<<<512, 128, 0, stream>>>(hbuf, bnsum, bnsumsq, N);
    k_bnfinal<<<1, 128, 0, stream>>>(bnsum, bnsumsq, ga, be, scale, shift, N);
    k_gemm_out<<<gG2, THREADS, 0, stream>>>(hbuf, scale, shift, Wb, bb, Yout, N);
  };

  layer(x,    Wa1, ba1, ga1, be1, Wb1, bb1, act1);
  layer(act1, Wa2, ba2, ga2, be2, Wb2, bb2, act2);

  k_zero<<<(Gn * 65 + THREADS - 1) / THREADS, THREADS, 0, stream>>>(pooled,
                                                      (long long)Gn * 64 + Gn);
  k_pool<<<gN64, THREADS, 0, stream>>>(act2, batch, pooled, cnt, n64);
  k_fc<<<1, ((Gn + 31) / 32) * 32, 0, stream>>>(pooled, cnt, Wfc, bfc, out, Gn);
}
